// GaussianVectorQuantizer_41953240547407
// MI455X (gfx1250) — matrix operand order repaired and checked by reading, not compiled, not run on hardware
//
#include <hip/hip_runtime.h>

typedef __bf16 bf16_t;
typedef __bf16 v16bf __attribute__((ext_vector_type(16)));
typedef __bf16 v8bf  __attribute__((ext_vector_type(8)));
typedef float  v8f   __attribute__((ext_vector_type(8)));

#define N_ROWS 65536   // B * NPTS
#define NDIM   64
#define BOOK   1024
#define WAVES  4
#define ROWS_PER_BLOCK 64   // 4 waves * 16 rows

// ---------------- prep: bf16 book, bf16 book^T, code norms, precision_q ----
__global__ void gvq_prep_kernel(const float* __restrict__ book,
                                const float* __restrict__ lpq,
                                bf16_t* __restrict__ book_bf,
                                bf16_t* __restrict__ bookT_bf,
                                float* __restrict__ bnorm,
                                float* __restrict__ pq_out) {
    int c = blockIdx.x * blockDim.x + threadIdx.x;
    if (c < BOOK) {
        float s = 0.f;
        #pragma unroll 8
        for (int d = 0; d < NDIM; ++d) {
            float v = book[c * NDIM + d];
            s += v * v;
            bf16_t b = (bf16_t)v;
            book_bf[c * NDIM + d]  = b;
            bookT_bf[d * BOOK + c] = b;
        }
        bnorm[c] = s;
    }
    if (c == 0) {
        pq_out[0] = 0.5f / fmaxf(__expf(lpq[0]), 1e-10f);
    }
}

// ---------------- fused: dist-GEMM + gumbel + flash-softmax + zq-GEMM ------
__global__ __launch_bounds__(128)
void gvq_main_kernel(const float* __restrict__ ze,
                     const float* __restrict__ lpq,
                     const float* __restrict__ u,
                     const bf16_t* __restrict__ book_bf,
                     const bf16_t* __restrict__ bookT_bf,
                     const float* __restrict__ bnorm,
                     float* __restrict__ zq_out,
                     float* __restrict__ logits_out) {
    extern __shared__ char smem[];
    bf16_t* sstash = (bf16_t*)smem;                                   // [WAVES][16][1024] bf16
    bf16_t* zshare = (bf16_t*)(smem + WAVES * 16 * 1024 * 2);         // [WAVES][16][64]  bf16
    float*  znorm  = (float*)(smem + WAVES * 16 * 1024 * 2 + WAVES * 16 * 64 * 2); // [WAVES][16]
    float*  Mlds   = znorm + WAVES * 16;                              // [WAVES][16]
    float*  Llds   = Mlds  + WAVES * 16;                              // [WAVES][16]

    const int tid  = threadIdx.x;
    const int wave = tid >> 5;
    const int lane = tid & 31;
    const int hl   = lane >> 4;      // half-wave: 0 or 1
    const int l16  = lane & 15;
    const int rowbase = blockIdx.x * ROWS_PER_BLOCK + wave * 16;

    const float pq = 0.5f / fmaxf(__expf(lpq[0]), 1e-10f);

    bf16_t* zb = zshare + wave * 16 * NDIM;
    bf16_t* ss = sstash + wave * 16 * BOOK;
    float*  zn = znorm  + wave * 16;
    float*  Mw = Mlds   + wave * 16;
    float*  Lw = Llds   + wave * 16;

    // ---- stage this wave's 16 z rows into LDS as bf16, compute row norms
    if (hl == 0) {
        const int r = l16;
        const float* zp = ze + (size_t)(rowbase + r) * NDIM;
        float s = 0.f;
        #pragma unroll 8
        for (int d = 0; d < NDIM; ++d) {
            float v = zp[d];
            s += v * v;
            zb[r * NDIM + d] = (bf16_t)v;
        }
        zn[r] = s;
    }
    __syncthreads();

    // ---- A-matrix (z tile) registers: constant for the entire code loop.
    // A as loaded here (contiguous k, same order as B): lane holds row l16; K elems {hl*16..+15}
    v8bf z00 = *(const v8bf*)(zb + l16 * NDIM +      hl * 16);
    v8bf z01 = *(const v8bf*)(zb + l16 * NDIM +  8 + hl * 16);
    v8bf z10 = *(const v8bf*)(zb + l16 * NDIM + 32 + hl * 16);
    v8bf z11 = *(const v8bf*)(zb + l16 * NDIM + 40 + hl * 16);
    v16bf a0 = __builtin_shufflevector(z00, z01, 0,1,2,3,4,5,6,7,8,9,10,11,12,13,14,15);
    v16bf a1 = __builtin_shufflevector(z10, z11, 0,1,2,3,4,5,6,7,8,9,10,11,12,13,14,15);

    float znr[8];
    #pragma unroll
    for (int e = 0; e < 8; ++e) znr[e] = zn[e + hl * 8];   // C-layout rows this lane owns

    float mrun[8];
    #pragma unroll
    for (int e = 0; e < 8; ++e) mrun[e] = -3.0e38f;

    // =========== Pass A: distances -> logits(out) + scores(LDS) + row max ==
    for (int cb = 0; cb < BOOK; cb += 16) {
        const int code = cb + l16;
        // B layout: lane = column(code); K dims {hl*16..+15}; 2nd WMMA dims +32
        const bf16_t* bp = book_bf + (size_t)code * NDIM + hl * 16;
        v16bf b0 = *(const v16bf*)bp;
        v16bf b1 = *(const v16bf*)(bp + 32);

        v8f S = {0.f, 0.f, 0.f, 0.f, 0.f, 0.f, 0.f, 0.f};
        S = __builtin_amdgcn_wmma_f32_16x16x32_bf16(false, a0, false, b0, (short)0, S, false, false);
        S = __builtin_amdgcn_wmma_f32_16x16x32_bf16(false, a1, false, b1, (short)0, S, false, false);

        const float bn = bnorm[code];
        const int pcb = (cb + 16 < BOOK) ? cb + 16 : cb;   // prefetch next u chunk

        #pragma unroll
        for (int e = 0; e < 8; ++e) {
            const int row = e + hl * 8;
            const size_t gr = (size_t)(rowbase + row);
            __builtin_prefetch(&u[gr * BOOK + pcb + l16], 0, 1);
            const float dot = S[e];
            const float logit = (2.0f * dot - znr[e] - bn) * pq;
            logits_out[gr * BOOK + code] = logit;
            const float uu = u[gr * BOOK + code];
            const float g = -__logf(-__logf(uu + 1e-10f) + 1e-10f);
            const float sval = (logit + g) * 2.0f;          // / TEMPERATURE(0.5)
            mrun[e] = fmaxf(mrun[e], sval);
            ss[row * BOOK + code] = (bf16_t)sval;
        }
    }

    // reduce row max across the 16 lanes holding each row (xor 1,2,4,8)
    #pragma unroll
    for (int off = 1; off < 16; off <<= 1) {
        #pragma unroll
        for (int e = 0; e < 8; ++e)
            mrun[e] = fmaxf(mrun[e], __shfl_xor(mrun[e], off, 32));
    }
    if (l16 == 0) {
        #pragma unroll
        for (int e = 0; e < 8; ++e) Mw[e + hl * 8] = mrun[e];
    }
    __syncthreads();

    // =========== Pass B: p = exp(s - M); zq_acc += p @ book; L += sum(p) ===
    const float Mrow = Mw[l16];    // A-layout: this lane holds row l16
    float lacc = 0.f;
    v8f acc0 = {0.f,0.f,0.f,0.f,0.f,0.f,0.f,0.f};
    v8f acc1 = acc0, acc2 = acc0, acc3 = acc0;

    for (int cb = 0; cb < BOOK; cb += 32) {
        // A (P tile): lane row l16, K codes {cb+hl*16..+15}
        v8bf slo = *(const v8bf*)(ss + l16 * BOOK + cb +     hl * 16);
        v8bf shi = *(const v8bf*)(ss + l16 * BOOK + cb + 8 + hl * 16);
        v16bf pa;
        #pragma unroll
        for (int i = 0; i < 8; ++i) {
            float p0 = __expf((float)slo[i] - Mrow);
            float p1 = __expf((float)shi[i] - Mrow);
            lacc += p0 + p1;
            pa[i]     = (bf16_t)p0;
            pa[i + 8] = (bf16_t)p1;
        }
        // B: lane = column(dim); K codes {cb+hl*16..+15}; book^T rows contiguous
        v16bf bt0 = *(const v16bf*)(bookT_bf + (size_t)( 0 + l16) * BOOK + cb + hl * 16);
        v16bf bt1 = *(const v16bf*)(bookT_bf + (size_t)(16 + l16) * BOOK + cb + hl * 16);
        v16bf bt2 = *(const v16bf*)(bookT_bf + (size_t)(32 + l16) * BOOK + cb + hl * 16);
        v16bf bt3 = *(const v16bf*)(bookT_bf + (size_t)(48 + l16) * BOOK + cb + hl * 16);
        acc0 = __builtin_amdgcn_wmma_f32_16x16x32_bf16(false, pa, false, bt0, (short)0, acc0, false, false);
        acc1 = __builtin_amdgcn_wmma_f32_16x16x32_bf16(false, pa, false, bt1, (short)0, acc1, false, false);
        acc2 = __builtin_amdgcn_wmma_f32_16x16x32_bf16(false, pa, false, bt2, (short)0, acc2, false, false);
        acc3 = __builtin_amdgcn_wmma_f32_16x16x32_bf16(false, pa, false, bt3, (short)0, acc3, false, false);
    }

    // softmax denominator: each row's codes split across lane pair {r, r+16}
    lacc += __shfl_xor(lacc, 16, 32);
    if (hl == 0) Lw[l16] = lacc;
    __syncthreads();

    // ---- write zq = acc / L  (C layout: VGPR e -> row e + hl*8, col l16)
    #pragma unroll
    for (int e = 0; e < 8; ++e) {
        const int row = e + hl * 8;
        const float linv = 1.0f / Lw[row];
        const size_t gr = (size_t)(rowbase + row);
        zq_out[gr * NDIM +  0 + l16] = acc0[e] * linv;
        zq_out[gr * NDIM + 16 + l16] = acc1[e] * linv;
        zq_out[gr * NDIM + 32 + l16] = acc2[e] * linv;
        zq_out[gr * NDIM + 48 + l16] = acc3[e] * linv;
    }
}

extern "C" void kernel_launch(void* const* d_in, const int* in_sizes, int n_in,
                              void* d_out, int out_size, void* d_ws, size_t ws_size,
                              hipStream_t stream) {
    const float* ze   = (const float*)d_in[0];
    const float* book = (const float*)d_in[1];
    const float* lpq  = (const float*)d_in[2];
    const float* u    = (const float*)d_in[3];

    float* out    = (float*)d_out;
    float* zq     = out;                         // 65536*64 floats
    float* pq_out = out + (size_t)N_ROWS * NDIM; // 1 float
    float* logits = pq_out + 1;                  // 65536*1024 floats

    bf16_t* book_bf  = (bf16_t*)d_ws;                          // 128 KB
    bf16_t* bookT_bf = (bf16_t*)((char*)d_ws + 131072);        // 128 KB
    float*  bnorm    = (float*)((char*)d_ws + 262144);         // 4 KB

    gvq_prep_kernel<<<(BOOK + 255) / 256, 256, 0, stream>>>(
        book, lpq, book_bf, bookT_bf, bnorm, pq_out);

    const size_t shmem = (size_t)WAVES * 16 * 1024 * 2   // score stash
                       + (size_t)WAVES * 16 * 64 * 2     // z tile
                       + (size_t)WAVES * 16 * 4 * 3;     // znorm, M, L
    gvq_main_kernel<<<N_ROWS / ROWS_PER_BLOCK, WAVES * 32, shmem, stream>>>(
        ze, lpq, u, book_bf, bookT_bf, bnorm, zq, logits);
}
